// BidirectionalMambaBlock_29695403885124
// MI455X (gfx1250) — compile-verified, matrix operand order checked by reading, not run on hardware
//
#include <hip/hip_runtime.h>
#include <hip/hip_bf16.h>

// ---------------- problem constants ----------------
#define BB   2
#define LL   1024
#define DM   768
#define DI   1536
#define DS   16
#define DTR  48

typedef __attribute__((ext_vector_type(16))) __bf16 bf16x16;
typedef __attribute__((ext_vector_type(8)))  float  v8f;

// ---------------------------------------------------------------------------
// Row LayerNorm (W columns, one block per row, 256 threads)
// ---------------------------------------------------------------------------
__global__ void ln_row_kernel(const float* __restrict__ src,
                              const float* __restrict__ g,
                              const float* __restrict__ bt,
                              float* __restrict__ dst, int W) {
  int row = blockIdx.x;
  const float* r = src + (size_t)row * W;
  float* o = dst + (size_t)row * W;
  __shared__ float red[256];
  float s = 0.f;
  for (int i = threadIdx.x; i < W; i += 256) s += r[i];
  red[threadIdx.x] = s; __syncthreads();
  for (int st = 128; st > 0; st >>= 1) {
    if (threadIdx.x < st) red[threadIdx.x] += red[threadIdx.x + st];
    __syncthreads();
  }
  float mean = red[0] / W; __syncthreads();
  float v = 0.f;
  for (int i = threadIdx.x; i < W; i += 256) { float d = r[i] - mean; v += d * d; }
  red[threadIdx.x] = v; __syncthreads();
  for (int st = 128; st > 0; st >>= 1) {
    if (threadIdx.x < st) red[threadIdx.x] += red[threadIdx.x + st];
    __syncthreads();
  }
  float rstd = rsqrtf(red[0] / W + 1e-5f);
  for (int i = threadIdx.x; i < W; i += 256)
    o[i] = (r[i] - mean) * rstd * g[i] + bt[i];
}

// ---------------------------------------------------------------------------
// Time-flip (B,L,W)
// ---------------------------------------------------------------------------
__global__ void flip_kernel(const float* __restrict__ src,
                            float* __restrict__ dst, int W) {
  int idx = blockIdx.x * blockDim.x + threadIdx.x;
  int total = BB * LL * W;
  if (idx >= total) return;
  int w = idx % W, row = idx / W;
  int t = row % LL, b = row / LL;
  dst[idx] = src[(size_t)(b * LL + (LL - 1 - t)) * W + w];
}

// ---------------------------------------------------------------------------
// Pack fp32 row-major A (MxK) -> bf16 WMMA A layout.
// ISA 16-bit A 16x32 tile: lane<16 holds M=lane, K={0..7,16..23};
// lane>=16 holds M=lane-16, K={8..15,24..31}.  Record = 16 bf16 per lane.
// dst index: ((mt*Ktiles + kt)*32 + lane)*16 + j
// ---------------------------------------------------------------------------
__global__ void pack_a_kernel(const float* __restrict__ src,
                              __bf16* __restrict__ dst,
                              int Mtiles, int Ktiles, int Kreal, int lda) {
  int tid = blockIdx.x * blockDim.x + threadIdx.x;
  int total = Mtiles * Ktiles * 32;
  if (tid >= total) return;
  int lane = tid & 31, tile = tid >> 5;
  int kt = tile % Ktiles, mt = tile / Ktiles;
  int m = mt * 16 + (lane & 15);
  int kb = (lane < 16) ? 0 : 8;
  __bf16* o = dst + (size_t)tile * 512 + (size_t)lane * 16;
#pragma unroll
  for (int j = 0; j < 16; ++j) {
    int k = kt * 32 + ((j < 8) ? (kb + j) : (16 + kb + (j - 8)));
    float v = (k < Kreal) ? src[(size_t)m * lda + k] : 0.f;
    o[j] = (__bf16)v;
  }
}

// Pack fp32 row-major B (KxN, ld = N) -> bf16 WMMA B layout (dual of A:
// lane holds one column n, same K mapping as A's per-lane K set).
__global__ void pack_b_kernel(const float* __restrict__ src,
                              __bf16* __restrict__ dst,
                              int Ntiles, int Ktiles, int Kreal, int ldb) {
  int tid = blockIdx.x * blockDim.x + threadIdx.x;
  int total = Ntiles * Ktiles * 32;
  if (tid >= total) return;
  int lane = tid & 31, tile = tid >> 5;
  int kt = tile % Ktiles, nt = tile / Ktiles;
  int n = nt * 16 + (lane & 15);
  int kb = (lane < 16) ? 0 : 8;
  __bf16* o = dst + (size_t)tile * 512 + (size_t)lane * 16;
#pragma unroll
  for (int j = 0; j < 16; ++j) {
    int k = kt * 32 + ((j < 8) ? (kb + j) : (16 + kb + (j - 8)));
    float v = (k < Kreal) ? src[(size_t)k * ldb + n] : 0.f;
    o[j] = (__bf16)v;
  }
}

// ---------------------------------------------------------------------------
// Register-blocked WMMA GEMM: one wave computes a BMxBN grid of 16x16 tiles.
// BM=4, BN=2: 8 accumulators (64 VGPRs) + 6 fragments (48 VGPRs) -- fits the
// register file with no spills; 8 WMMAs per 6 fragment loads per K-step.
// Addressing is strength-reduced: one base pointer per operand advanced by
// 1024 B per K-step, tile offsets (i*Kt*512 elements) are loop-invariant.
// C layout per ISA: lane<16 -> N=lane, rows M=m0..m0+7; lane>=16 -> rows +8.
// Requires Mt % BM == 0 and Nt % BN == 0.
// ---------------------------------------------------------------------------
template <int BM, int BN>
__global__ __launch_bounds__(128, 1)
void wmma_gemm_kernel(const __bf16* __restrict__ pa,
                      const __bf16* __restrict__ pb,
                      float* __restrict__ C,
                      int Mt, int Nt, int Kt, int ldc,
                      const float* __restrict__ bias) {
  int wave = (blockIdx.x * blockDim.x + threadIdx.x) >> 5;
  int nbm = Mt / BM;
  if (wave >= nbm * (Nt / BN)) return;      // wave-uniform exit, EXEC stays full
  int lane = threadIdx.x & 31;
  int mt0 = (wave % nbm) * BM;
  int nt0 = (wave / nbm) * BN;
  const size_t tstride = (size_t)Kt * 512;  // elements between adjacent M/N tiles
  const __bf16* abase = pa + (size_t)mt0 * tstride + (size_t)lane * 16;
  const __bf16* bbase = pb + (size_t)nt0 * tstride + (size_t)lane * 16;

  v8f acc[BM][BN];
#pragma unroll
  for (int i = 0; i < BM; ++i)
#pragma unroll
    for (int j = 0; j < BN; ++j) acc[i][j] = {};

  for (int kt = 0; kt < Kt; ++kt) {
    bf16x16 a[BM], b[BN];
#pragma unroll
    for (int i = 0; i < BM; ++i)
      a[i] = *(const bf16x16*)(abase + (size_t)i * tstride);
#pragma unroll
    for (int j = 0; j < BN; ++j)
      b[j] = *(const bf16x16*)(bbase + (size_t)j * tstride);
    abase += 512; bbase += 512;
#pragma unroll
    for (int i = 0; i < BM; ++i)
#pragma unroll
      for (int j = 0; j < BN; ++j)
        acc[i][j] = __builtin_amdgcn_wmma_f32_16x16x32_bf16(
            false, a[i], false, b[j], (short)0, acc[i][j], false, false);
  }

  int nlo = lane & 15;
  int mof = (lane < 16) ? 0 : 8;
#pragma unroll
  for (int i = 0; i < BM; ++i) {
    int m0 = (mt0 + i) * 16 + mof;
#pragma unroll
    for (int j = 0; j < BN; ++j) {
      int ncol = (nt0 + j) * 16 + nlo;
      float bv = bias ? bias[ncol] : 0.f;
#pragma unroll
      for (int r = 0; r < 8; ++r)
        C[(size_t)(m0 + r) * ldc + ncol] = acc[i][j][r] + bv;
    }
  }
}

// ---------------------------------------------------------------------------
// Causal depthwise conv (k=4) + bias + SiLU.  xi = xz[:, :, 0:DI].
// ---------------------------------------------------------------------------
__global__ void conv_silu_kernel(const float* __restrict__ xz,
                                 const float* __restrict__ cw,
                                 const float* __restrict__ cb,
                                 float* __restrict__ xc) {
  int idx = blockIdx.x * blockDim.x + threadIdx.x;
  int total = BB * LL * DI;
  if (idx >= total) return;
  int d = idx % DI, row = idx / DI;
  int t = row % LL, b = row / LL;
  float acc = cb[d];
#pragma unroll
  for (int i = 0; i < 4; ++i) {
    int ts = t - 3 + i;
    if (ts >= 0) acc += cw[d * 4 + i] * xz[(size_t)(b * LL + ts) * (2 * DI) + d];
  }
  xc[idx] = acc / (1.f + __expf(-acc));
}

// ---------------------------------------------------------------------------
// Selective scan. y buffer holds dt_pre on entry; y[b,t,d] written in place
// after the step consumes dt_pre[b,t,d].  B/C (32 floats per step, shared by
// all 1536 channels of a batch) staged in LDS in 32-step chunks.
// grid = BB*DI/256 blocks; each block covers 256 channels of one batch.
// ---------------------------------------------------------------------------
__global__ void scan_kernel(float* __restrict__ y,
                            const float* __restrict__ dbc,
                            const float* __restrict__ xc,
                            const float* __restrict__ Alog,
                            const float* __restrict__ dtb,
                            const float* __restrict__ Dp) {
  __shared__ float sBC[32 * 32];            // [tt][c]: c<16 -> B, c>=16 -> C
  int ch = blockIdx.x * blockDim.x + threadIdx.x;  // 0..3071
  int b = ch / DI, d = ch % DI;
  float Arow[DS];
#pragma unroll
  for (int n = 0; n < DS; ++n) Arow[n] = -__expf(Alog[d * DS + n]);
  float dtbv = dtb[d], Dv = Dp[d];
  float h[DS];
#pragma unroll
  for (int n = 0; n < DS; ++n) h[n] = 0.f;

  for (int t0 = 0; t0 < LL; t0 += 32) {
    __syncthreads();
    for (int e = threadIdx.x; e < 32 * 32; e += 256) {
      int tt = e >> 5, c = e & 31;
      sBC[e] = dbc[(size_t)(b * LL + t0 + tt) * 80 + DTR + c];
    }
    __syncthreads();
    for (int tt = 0; tt < 32; ++tt) {
      size_t idx = (size_t)(b * LL + t0 + tt) * DI + d;
      float xx = y[idx] + dtbv;                       // dt_pre + dt bias
      float dt = (xx > 20.f) ? xx : log1pf(__expf(xx));  // softplus
      float u = xc[idx];
      float acc = 0.f;
#pragma unroll
      for (int n = 0; n < DS; ++n) {
        float dA = __expf(dt * Arow[n]);
        h[n] = dA * h[n] + dt * sBC[tt * 32 + n] * u;
        acc += h[n] * sBC[tt * 32 + 16 + n];
      }
      y[idx] = acc + u * Dv;
    }
  }
}

// y *= silu(z), z = xz[:, :, DI:2*DI]
__global__ void gate_kernel(float* __restrict__ y, const float* __restrict__ xz) {
  int idx = blockIdx.x * blockDim.x + threadIdx.x;
  int total = BB * LL * DI;
  if (idx >= total) return;
  int d = idx % DI, row = idx / DI;
  float z = xz[(size_t)row * (2 * DI) + DI + d];
  y[idx] *= z / (1.f + __expf(-z));
}

// Write a direction's (2048x768) output into cat (2048x1536) at colOff,
// flipping time for the backward direction.
__global__ void cat_copy_kernel(const float* __restrict__ src,
                                float* __restrict__ cat, int flip, int colOff) {
  int idx = blockIdx.x * blockDim.x + threadIdx.x;
  int total = BB * LL * DM;
  if (idx >= total) return;
  int w = idx % DM, row = idx / DM;
  int t = row % LL, b = row / LL;
  int sr = flip ? (b * LL + (LL - 1 - t)) : row;
  cat[(size_t)row * (2 * DM) + colOff + w] = src[(size_t)sr * DM + w];
}

// out = x + gelu_tanh(LN(fused_pre))   (fused_pre already has fuse_b added)
__global__ void fuse_epilogue_kernel(const float* __restrict__ fp,
                                     const float* __restrict__ g,
                                     const float* __restrict__ bt,
                                     const float* __restrict__ x,
                                     float* __restrict__ out) {
  int row = blockIdx.x;
  const float* r = fp + (size_t)row * DM;
  __shared__ float red[256];
  float s = 0.f;
  for (int i = threadIdx.x; i < DM; i += 256) s += r[i];
  red[threadIdx.x] = s; __syncthreads();
  for (int st = 128; st > 0; st >>= 1) {
    if (threadIdx.x < st) red[threadIdx.x] += red[threadIdx.x + st];
    __syncthreads();
  }
  float mean = red[0] / DM; __syncthreads();
  float v = 0.f;
  for (int i = threadIdx.x; i < DM; i += 256) { float d = r[i] - mean; v += d * d; }
  red[threadIdx.x] = v; __syncthreads();
  for (int st = 128; st > 0; st >>= 1) {
    if (threadIdx.x < st) red[threadIdx.x] += red[threadIdx.x + st];
    __syncthreads();
  }
  float rstd = rsqrtf(red[0] / DM + 1e-5f);
  for (int i = threadIdx.x; i < DM; i += 256) {
    float vv = (r[i] - mean) * rstd * g[i] + bt[i];
    float ge = 0.5f * vv * (1.f + tanhf(0.7978845608f * (vv + 0.044715f * vv * vv * vv)));
    out[(size_t)row * DM + i] = x[(size_t)row * DM + i] + ge;
  }
}

// ---------------------------------------------------------------------------
// Orchestration
// ---------------------------------------------------------------------------
static inline int cdiv_i(int a, int b) { return (a + b - 1) / b; }

extern "C" void kernel_launch(void* const* d_in, const int* in_sizes, int n_in,
                              void* d_out, int out_size, void* d_ws, size_t ws_size,
                              hipStream_t stream) {
  (void)in_sizes; (void)n_in; (void)out_size; (void)ws_size;
  const float* x      = (const float*)d_in[0];
  const float* pn_g   = (const float*)d_in[1];
  const float* pn_b   = (const float*)d_in[2];
  const float* fuse_W = (const float*)d_in[3];
  const float* fuse_b = (const float*)d_in[4];
  const float* fl_g   = (const float*)d_in[5];
  const float* fl_b   = (const float*)d_in[6];

  char* p = (char*)d_ws;
  auto take = [&](size_t bytes) {
    char* r = p; p += (bytes + 255) & ~(size_t)255; return (void*)r;
  };
  float*  xn   = (float*)take((size_t)BB * LL * DM * 4);
  float*  xnr  = (float*)take((size_t)BB * LL * DM * 4);
  float*  xz   = (float*)take((size_t)BB * LL * 2 * DI * 4);
  float*  xc   = (float*)take((size_t)BB * LL * DI * 4);
  float*  dbc  = (float*)take((size_t)BB * LL * 80 * 4);
  float*  dty  = (float*)take((size_t)BB * LL * DI * 4);   // dt_pre -> y (in place)
  float*  dout = (float*)take((size_t)BB * LL * DM * 4);
  float*  cat  = (float*)take((size_t)BB * LL * 2 * DM * 4);
  float*  fpre = (float*)take((size_t)BB * LL * DM * 4);
  __bf16* pA   = (__bf16*)take((size_t)128 * 48 * 512 * 2); // max Mt*Kt tiles
  __bf16* pB   = (__bf16*)take((size_t)192 * 24 * 512 * 2); // max Nt*Kt tiles

  const int ROWS = BB * LL;                 // 2048 rows, Mt = 128 everywhere

  ln_row_kernel<<<ROWS, 256, 0, stream>>>(x, pn_g, pn_b, xn, DM);
  flip_kernel<<<cdiv_i(ROWS * DM, 256), 256, 0, stream>>>(xn, xnr, DM);

  for (int dir = 0; dir < 2; ++dir) {
    int base = 7 + dir * 9;
    const float* inW   = (const float*)d_in[base + 0];
    const float* convW = (const float*)d_in[base + 1];
    const float* convb = (const float*)d_in[base + 2];
    const float* xpW   = (const float*)d_in[base + 3];
    const float* dtW   = (const float*)d_in[base + 4];
    const float* dtb   = (const float*)d_in[base + 5];
    const float* Alog  = (const float*)d_in[base + 6];
    const float* Dp    = (const float*)d_in[base + 7];
    const float* outW  = (const float*)d_in[base + 8];
    const float* xin = dir ? xnr : xn;

    // xz = xin @ inW : (2048x768)x(768x3072), Mt=128 Nt=192 Kt=24
    pack_a_kernel<<<cdiv_i(128 * 24 * 32, 256), 256, 0, stream>>>(xin, pA, 128, 24, DM, DM);
    pack_b_kernel<<<cdiv_i(192 * 24 * 32, 256), 256, 0, stream>>>(inW, pB, 192, 24, DM, 2 * DI);
    {
      int waves = (128 / 4) * (192 / 2);    // 3072
      wmma_gemm_kernel<4, 2><<<cdiv_i(waves * 32, 128), 128, 0, stream>>>(
          pA, pB, xz, 128, 192, 24, 2 * DI, nullptr);
    }

    conv_silu_kernel<<<cdiv_i(ROWS * DI, 256), 256, 0, stream>>>(xz, convW, convb, xc);

    // dbc = xc @ xpW : (2048x1536)x(1536x80), Mt=128 Nt=5 Kt=48
    pack_a_kernel<<<cdiv_i(128 * 48 * 32, 256), 256, 0, stream>>>(xc, pA, 128, 48, DI, DI);
    pack_b_kernel<<<cdiv_i(5 * 48 * 32, 256), 256, 0, stream>>>(xpW, pB, 5, 48, DI, 80);
    {
      int waves = (128 / 4) * 5;            // 160 (BN=1: Nt=5 not 2-divisible)
      wmma_gemm_kernel<4, 1><<<cdiv_i(waves * 32, 128), 128, 0, stream>>>(
          pA, pB, dbc, 128, 5, 48, 80, nullptr);
    }

    // dt_pre = dbc[:, :48] @ dtW : K=48 padded to 64, Mt=128 Nt=96 Kt=2
    pack_a_kernel<<<cdiv_i(128 * 2 * 32, 256), 256, 0, stream>>>(dbc, pA, 128, 2, DTR, 80);
    pack_b_kernel<<<cdiv_i(96 * 2 * 32, 256), 256, 0, stream>>>(dtW, pB, 96, 2, DTR, DI);
    {
      int waves = (128 / 4) * (96 / 2);     // 1536
      wmma_gemm_kernel<4, 2><<<cdiv_i(waves * 32, 128), 128, 0, stream>>>(
          pA, pB, dty, 128, 96, 2, DI, nullptr);
    }

    // selective scan (dt_pre consumed, y produced in place) + gate
    scan_kernel<<<(BB * DI) / 256, 256, 0, stream>>>(dty, dbc, xc, Alog, dtb, Dp);
    gate_kernel<<<cdiv_i(ROWS * DI, 256), 256, 0, stream>>>(dty, xz);

    // dir_out = y @ outW : (2048x1536)x(1536x768), Mt=128 Nt=48 Kt=48
    pack_a_kernel<<<cdiv_i(128 * 48 * 32, 256), 256, 0, stream>>>(dty, pA, 128, 48, DI, DI);
    pack_b_kernel<<<cdiv_i(48 * 48 * 32, 256), 256, 0, stream>>>(outW, pB, 48, 48, DI, DM);
    {
      int waves = (128 / 4) * (48 / 2);     // 768
      wmma_gemm_kernel<4, 2><<<cdiv_i(waves * 32, 128), 128, 0, stream>>>(
          pA, pB, dout, 128, 48, 48, DM, nullptr);
    }

    cat_copy_kernel<<<cdiv_i(ROWS * DM, 256), 256, 0, stream>>>(dout, cat, dir, dir * DM);
  }

  // fused_pre = cat @ fuse_W + fuse_b : (2048x1536)x(1536x768)
  pack_a_kernel<<<cdiv_i(128 * 48 * 32, 256), 256, 0, stream>>>(cat, pA, 128, 48, 2 * DM, 2 * DM);
  pack_b_kernel<<<cdiv_i(48 * 48 * 32, 256), 256, 0, stream>>>(fuse_W, pB, 48, 48, 2 * DM, DM);
  {
    int waves = (128 / 4) * (48 / 2);       // 768
    wmma_gemm_kernel<4, 2><<<cdiv_i(waves * 32, 128), 128, 0, stream>>>(
        pA, pB, fpre, 128, 48, 48, DM, fuse_b);
  }

  fuse_epilogue_kernel<<<ROWS, 256, 0, stream>>>(fpre, fl_g, fl_b, x, (float*)d_out);
}
